// shared_multimodality_roi_71665824301535
// MI455X (gfx1250) — matrix operand order repaired and checked by reading, not compiled, not run on hardware
//
#include <hip/hip_runtime.h>
#include <math.h>

// ---------------- CDNA5 WMMA types ----------------
typedef __attribute__((ext_vector_type(16))) _Float16 v16h;
typedef __attribute__((ext_vector_type(8)))  float    v8f;

// =====================================================================
// Fragment layouts used by these kernels (wave32, contiguous k order):
//   A (16x32 f16): lane l -> row M=l&15; element e holds
//                  K = 16*(l>>4) + e
//   B (32x16 f16): lane l -> col N=l&15; element jj holds K = 16*(l>>4)+jj
//   C/D (16x16 f32): VGPR r -> M = r + 8*(l>>4), N = l&15
// =====================================================================

// ---------------------------------------------------------------------
// Per-wave implicit GEMM (one wave/block, MT M-tiles reuse one B fragment).
// Used for Cout < 256. Branch-free hot loop via clamping + value masking.
// ---------------------------------------------------------------------
template<int KS, int MT>
__global__ void __launch_bounds__(32)
gemm_wmma_kernel(const float* __restrict__ Wt, const float* __restrict__ bias,
                 const float* __restrict__ X, float* __restrict__ Y,
                 int Cout, int Cin, int pad,
                 int Bn, int D, int H, int Wd,
                 int tilesM, int coOff, int Ctot, float scale, int doLeaky)
{
    const int lane  = threadIdx.x;
    const int tile  = blockIdx.x;
    const int tmg   = tile % tilesM;       // M-group (MT*16 rows)
    const int tn    = tile / tilesM;
    const int P     = D * H * Wd;
    const int KV    = KS * KS * KS;
    const int K     = Cin * KV;
    const int khalf = lane >> 4;
    const int lan15 = lane & 15;
    const int m0    = tmg * (16 * MT);
    const int Ncols = Bn * P;

    const int  nB    = tn * 16 + lan15;
    const bool colOK = nB < Ncols;
    const int  nBc   = colOK ? nB : (Ncols - 1);   // clamp: loads valid, store guarded
    const int  b     = nBc / P;
    const int  p     = nBc % P;
    int od = 0, oh = 0, ow = 0;
    if (KS == 3) {
        od = p / (H * Wd);
        int r = p % (H * Wd);
        oh = r / Wd; ow = r % Wd;
    }
    const size_t HW = (size_t)H * Wd;

    const float* arow[MT];
#pragma unroll
    for (int mt = 0; mt < MT; ++mt) {
        int mr = m0 + mt * 16 + lan15;
        if (mr > Cout - 1) mr = Cout - 1;
        arow[mt] = Wt + (size_t)mr * K;
    }
    const float* xcol = X + (size_t)b * Cin * P;   // KS==1: element k at xcol + k*P + p

    v8f acc[MT];
    {
        v8f z = {};
#pragma unroll
        for (int mt = 0; mt < MT; ++mt) acc[mt] = z;
    }

    const int Kmain = K & ~31;
    int kb_ci = 0, kb_t = 0;               // running im2col decode (KS==3)
    if (KS == 3) { int ks0 = 16 * khalf; kb_ci = ks0 / KV; kb_t = ks0 % KV; }

    for (int k0 = 0; k0 < Kmain; k0 += 32) {
        v16h bf;
        if (KS == 1) {
            const float* xk = xcol + (size_t)(k0 + 16 * khalf) * P + p;
            __builtin_prefetch(xk + (size_t)32 * P, 0, 1);
#pragma unroll
            for (int jj = 0; jj < 16; ++jj)
                bf[jj] = (_Float16)xk[(size_t)jj * P];
        } else {
            int ci = kb_ci, t = kb_t;
#pragma unroll
            for (int jj = 0; jj < 16; ++jj) {
                int kd = t / 9, t2 = t - kd * 9, kh = t2 / 3, kw = t2 - kh * 3;
                int id = od + kd - pad, ih = oh + kh - pad, iw = ow + kw - pad;
                bool inb = ((unsigned)id < (unsigned)D) & ((unsigned)ih < (unsigned)H) &
                           ((unsigned)iw < (unsigned)Wd);
                int idc = inb ? id : 0, ihc = inb ? ih : 0, iwc = inb ? iw : 0;
                float v = X[(((size_t)b * Cin + ci) * D + idc) * HW + (size_t)ihc * Wd + iwc];
                bf[jj] = (_Float16)(inb ? v : 0.f);
                if (++t == KV) { t = 0; ++ci; }
            }
            t += 16; if (t >= KV) { t -= KV; ++ci; }
            kb_ci = ci; kb_t = t;
        }
#pragma unroll
        for (int mt = 0; mt < MT; ++mt) {
            v16h af;
#pragma unroll
            for (int i = 0; i < 8; ++i) {
                const int k = k0 + 16 * khalf + 2 * i;
                const float2 w2 = *(const float2*)(arow[mt] + k);  // K even whenever Kmain>0
                af[2 * i]     = (_Float16)w2.x;
                af[2 * i + 1] = (_Float16)w2.y;
            }
            acc[mt] = __builtin_amdgcn_wmma_f32_16x16x32_f16(false, af, false, bf,
                                                             (short)0, acc[mt], false, false);
        }
    }

    if (Kmain < K) {   // single guarded remainder chunk
        const int k0 = Kmain;
        v16h bf;
#pragma unroll
        for (int jj = 0; jj < 16; ++jj) {
            const int k  = k0 + 16 * khalf + jj;
            const int kc = (k < K) ? k : (K - 1);
            float v;
            if (KS == 1) {
                v = xcol[(size_t)kc * P + p];
            } else {
                int ci = kc / KV, t = kc % KV;
                int kd = t / 9, t2 = t - kd * 9, kh = t2 / 3, kw = t2 - kh * 3;
                int id = od + kd - pad, ih = oh + kh - pad, iw = ow + kw - pad;
                bool inb = ((unsigned)id < (unsigned)D) & ((unsigned)ih < (unsigned)H) &
                           ((unsigned)iw < (unsigned)Wd);
                int idc = inb ? id : 0, ihc = inb ? ih : 0, iwc = inb ? iw : 0;
                v = X[(((size_t)b * Cin + ci) * D + idc) * HW + (size_t)ihc * Wd + iwc];
                v = inb ? v : 0.f;
            }
            bf[jj] = (_Float16)((k < K) ? v : 0.f);
        }
#pragma unroll
        for (int mt = 0; mt < MT; ++mt) {
            v16h af;
#pragma unroll
            for (int e = 0; e < 16; ++e) {
                const int k  = k0 + 16 * khalf + e;
                const int kc = (k < K) ? k : (K - 1);
                const float w = arow[mt][kc];
                af[e] = (_Float16)((k < K) ? w : 0.f);
            }
            acc[mt] = __builtin_amdgcn_wmma_f32_16x16x32_f16(false, af, false, bf,
                                                             (short)0, acc[mt], false, false);
        }
    }

    if (!colOK) return;
#pragma unroll
    for (int mt = 0; mt < MT; ++mt) {
#pragma unroll
        for (int r = 0; r < 8; ++r) {
            const int m = m0 + mt * 16 + r + 8 * khalf;
            if (m >= Cout) continue;
            float v = acc[mt][r];
            if (bias) v += bias[m];
            v *= scale;
            if (doLeaky) v = (v >= 0.f) ? v : 0.2f * v;
            Y[((size_t)b * Ctot + coOff + m) * P + p] = v;
        }
    }
}

// ---------------------------------------------------------------------
// 8-wave block GEMM with LDS-shared B operand (Cout >= 256).
// All 8 waves share one N-tile; the 32x16 B tile is staged cooperatively
// into LDS (f16, double-buffered, one barrier per K-chunk) and consumed
// by every wave -> 8x fewer global B gathers. Each wave covers 32 rows
// (MT=2), so a block covers 256 output rows.
// ---------------------------------------------------------------------
template<int KS>
__global__ void __launch_bounds__(256)
gemm_wmma_lds_kernel(const float* __restrict__ Wt, const float* __restrict__ bias,
                     const float* __restrict__ X, float* __restrict__ Y,
                     int Cout, int Cin, int pad,
                     int Bn, int D, int H, int Wd,
                     int tilesMB, int coOff, int Ctot, float scale, int doLeaky)
{
    constexpr int MT  = 2;
    constexpr int WPB = 8;
    const int tid   = threadIdx.x;
    const int wave  = tid >> 5;
    const int lane  = tid & 31;
    const int tile  = blockIdx.x;
    const int tmb   = tile % tilesMB;       // 256-row super-group
    const int tn    = tile / tilesMB;
    const int P     = D * H * Wd;
    const int KV    = KS * KS * KS;
    const int K     = Cin * KV;
    const int khalf = lane >> 4;
    const int lan15 = lane & 15;
    const int m0    = (tmb * WPB + wave) * (16 * MT);
    const int Ncols = Bn * P;

    // store-side column decode (per lane)
    const int  nB    = tn * 16 + lan15;
    const bool colOK = nB < Ncols;
    const int  nBc   = colOK ? nB : (Ncols - 1);
    const int  b     = nBc / P;
    const int  p     = nBc % P;

    // staging assignment: this thread owns column scol, rows sk and sk+16
    const int scol = tid & 15;
    const int sk   = tid >> 4;              // 0..15
    const int snB  = tn * 16 + scol;
    const int snBc = (snB < Ncols) ? snB : (Ncols - 1);
    const int sbb  = snBc / P;
    const int spp  = snBc % P;
    int sod = 0, soh = 0, sow = 0;
    if (KS == 3) {
        sod = spp / (H * Wd);
        int r = spp % (H * Wd);
        soh = r / Wd; sow = r % Wd;
    }
    const size_t HW = (size_t)H * Wd;
    const float* sxcol = X + (size_t)sbb * Cin * P;

    __shared__ _Float16 bufB[2][16][32];    // [buffer][col][k] f16

    const float* arow[MT];
#pragma unroll
    for (int mt = 0; mt < MT; ++mt) {
        int mr = m0 + mt * 16 + lan15;
        if (mr > Cout - 1) mr = Cout - 1;
        arow[mt] = Wt + (size_t)mr * K;
    }

    v8f acc[MT];
    {
        v8f z = {};
#pragma unroll
        for (int mt = 0; mt < MT; ++mt) acc[mt] = z;
    }

    // KS==3 running im2col decode for the two staged k-streams (sk, sk+16)
    int ci0 = 0, t0 = 0, ci1 = 0, t1 = 0;
    if (KS == 3) {
        ci0 = sk / KV;        t0 = sk % KV;
        ci1 = (sk + 16) / KV; t1 = (sk + 16) % KV;
    }
    auto b3val = [&](int ci, int t) -> float {
        int kd = t / 9, t2 = t - kd * 9, kh = t2 / 3, kw = t2 - kh * 3;
        int id = sod + kd - pad, ih = soh + kh - pad, iw = sow + kw - pad;
        bool inb = ((unsigned)id < (unsigned)D) & ((unsigned)ih < (unsigned)H) &
                   ((unsigned)iw < (unsigned)Wd);
        int idc = inb ? id : 0, ihc = inb ? ih : 0, iwc = inb ? iw : 0;
        float v = X[(((size_t)sbb * Cin + ci) * D + idc) * HW + (size_t)ihc * Wd + iwc];
        return inb ? v : 0.f;
    };

    const int Kmain = K & ~31;
    int par = 0;
    for (int k0 = 0; k0 < Kmain; k0 += 32) {
        // ---- cooperative staging: 2 elements per thread ----
        if (KS == 1) {
            const float* base = sxcol + (size_t)k0 * P + spp;
            float v0 = base[(size_t)sk * P];
            float v1 = base[(size_t)(sk + 16) * P];
            __builtin_prefetch(base + (size_t)(sk + 32) * P, 0, 1);
            bufB[par][scol][sk]      = (_Float16)v0;
            bufB[par][scol][sk + 16] = (_Float16)v1;
        } else {
            float v0 = b3val(ci0, t0);
            float v1 = b3val(ci1, t1);
            bufB[par][scol][sk]      = (_Float16)v0;
            bufB[par][scol][sk + 16] = (_Float16)v1;
            t0 += 32; if (t0 >= KV) { t0 -= KV; ++ci0; } if (t0 >= KV) { t0 -= KV; ++ci0; }
            t1 += 32; if (t1 >= KV) { t1 -= KV; ++ci1; } if (t1 >= KV) { t1 -= KV; ++ci1; }
        }
        __syncthreads();   // double buffer -> single barrier per chunk
        // ---- consume shared B + per-wave weights ----
        v16h bf;
#pragma unroll
        for (int jj = 0; jj < 16; ++jj)
            bf[jj] = bufB[par][lan15][16 * khalf + jj];
#pragma unroll
        for (int mt = 0; mt < MT; ++mt) {
            v16h af;
#pragma unroll
            for (int i = 0; i < 8; ++i) {
                const int k = k0 + 16 * khalf + 2 * i;
                const float2 w2 = *(const float2*)(arow[mt] + k);
                af[2 * i]     = (_Float16)w2.x;
                af[2 * i + 1] = (_Float16)w2.y;
            }
            acc[mt] = __builtin_amdgcn_wmma_f32_16x16x32_f16(false, af, false, bf,
                                                             (short)0, acc[mt], false, false);
        }
        par ^= 1;
    }

    if (Kmain < K) {   // guarded remainder chunk (same staging scheme + k<K mask)
        const int k0 = Kmain;
        {
            const int ka = k0 + sk, kb2 = k0 + sk + 16;
            float v0, v1;
            if (KS == 1) {
                const int kac = (ka  < K) ? ka  : (K - 1);
                const int kbc = (kb2 < K) ? kb2 : (K - 1);
                v0 = sxcol[(size_t)kac * P + spp];
                v1 = sxcol[(size_t)kbc * P + spp];
            } else {
                const int kac = (ka  < K) ? ka  : (K - 1);
                const int kbc = (kb2 < K) ? kb2 : (K - 1);
                v0 = b3val(kac / KV, kac % KV);
                v1 = b3val(kbc / KV, kbc % KV);
            }
            bufB[par][scol][sk]      = (_Float16)((ka  < K) ? v0 : 0.f);
            bufB[par][scol][sk + 16] = (_Float16)((kb2 < K) ? v1 : 0.f);
        }
        __syncthreads();
        v16h bf;
#pragma unroll
        for (int jj = 0; jj < 16; ++jj)
            bf[jj] = bufB[par][lan15][16 * khalf + jj];
#pragma unroll
        for (int mt = 0; mt < MT; ++mt) {
            v16h af;
#pragma unroll
            for (int e = 0; e < 16; ++e) {
                const int k  = k0 + 16 * khalf + e;
                const int kc = (k < K) ? k : (K - 1);
                const float w = arow[mt][kc];
                af[e] = (_Float16)((k < K) ? w : 0.f);
            }
            acc[mt] = __builtin_amdgcn_wmma_f32_16x16x32_f16(false, af, false, bf,
                                                             (short)0, acc[mt], false, false);
        }
    }

    // ---- epilogue ----
    if (!colOK) return;
#pragma unroll
    for (int mt = 0; mt < MT; ++mt) {
#pragma unroll
        for (int r = 0; r < 8; ++r) {
            const int m = m0 + mt * 16 + r + 8 * khalf;
            if (m >= Cout) continue;
            float v = acc[mt][r];
            if (bias) v += bias[m];
            v *= scale;
            if (doLeaky) v = (v >= 0.f) ? v : 0.2f * v;
            Y[((size_t)b * Ctot + coOff + m) * P + p] = v;
        }
    }
}

// ---------------- depthwise 3x3x3 conv, pad=1, + bias (memory-bound) ----------------
__global__ void dwconv3_kernel(const float* __restrict__ x, const float* __restrict__ w,
                               const float* __restrict__ bias, float* __restrict__ y,
                               int Bn, int C, int D, int H, int Wd)
{
    size_t idx = (size_t)blockIdx.x * blockDim.x + threadIdx.x;
    size_t total = (size_t)Bn * C * D * H * Wd;
    if (idx >= total) return;
    int iw = idx % Wd; size_t t = idx / Wd;
    int ih = t % H; t /= H;
    int id = t % D; t /= D;
    int c  = t % C;
    int bb = t / C;
    const float* wc = w + c * 27;
    const float* xc = x + ((size_t)bb * C + c) * D * H * Wd;
    float acc = bias[c];
    for (int kd = 0; kd < 3; ++kd) {
        int zd = id + kd - 1;
        bool okd = (unsigned)zd < (unsigned)D;
        int zdc = okd ? zd : 0;
        for (int kh = 0; kh < 3; ++kh) {
            int zh = ih + kh - 1;
            bool okh = okd & ((unsigned)zh < (unsigned)H);
            int zhc = ((unsigned)zh < (unsigned)H) ? zh : 0;
            for (int kw = 0; kw < 3; ++kw) {
                int zw = iw + kw - 1;
                bool ok = okh & ((unsigned)zw < (unsigned)Wd);
                int zwc = ((unsigned)zw < (unsigned)Wd) ? zw : 0;
                float v = xc[((size_t)zdc * H + zhc) * Wd + zwc];
                acc += wc[(kd * 3 + kh) * 3 + kw] * (ok ? v : 0.f);
            }
        }
    }
    y[idx] = acc;
}

// ---------------- 2x2x2 max pool ----------------
__global__ void maxpool2_kernel(const float* __restrict__ x, float* __restrict__ y,
                                int BC, int D, int H, int Wd)
{
    int Do = D >> 1, Ho = H >> 1, Wo = Wd >> 1;
    size_t idx = (size_t)blockIdx.x * blockDim.x + threadIdx.x;
    size_t total = (size_t)BC * Do * Ho * Wo;
    if (idx >= total) return;
    int wo = idx % Wo; size_t t = idx / Wo;
    int ho = t % Ho; t /= Ho;
    int dd = t % Do;
    int bc = t / Do;
    const float* xi = x + (((size_t)bc * D + dd * 2) * H + ho * 2) * Wd + wo * 2;
    float m = -INFINITY;
    for (int a = 0; a < 2; ++a)
        for (int b2 = 0; b2 < 2; ++b2)
            for (int c2 = 0; c2 < 2; ++c2)
                m = fmaxf(m, xi[((size_t)a * H + b2) * Wd + c2]);
    y[idx] = m;
}

// ---------------- InstanceNorm3d (biased var), fused leaky, concat-offset write ----------------
__global__ void inorm_kernel(const float* __restrict__ x, float* __restrict__ y,
                             int C, int P, int coOff, int Ctot, int doLeaky)
{
    int bc = blockIdx.x;           // b*C + c
    int c  = bc % C;
    int b  = bc / C;
    const float* xi = x + (size_t)bc * P;
    float* yo = y + ((size_t)b * Ctot + coOff + c) * P;
    __shared__ float s1[256], s2[256];
    float s = 0.f, ss = 0.f;
    for (int i = threadIdx.x; i < P; i += blockDim.x) { float v = xi[i]; s += v; ss += v * v; }
    s1[threadIdx.x] = s; s2[threadIdx.x] = ss;
    __syncthreads();
    for (int o = 128; o > 0; o >>= 1) {
        if ((int)threadIdx.x < o) { s1[threadIdx.x] += s1[threadIdx.x + o]; s2[threadIdx.x] += s2[threadIdx.x + o]; }
        __syncthreads();
    }
    float mean = s1[0] / P;
    float var  = s2[0] / P - mean * mean;
    float inv  = rsqrtf(var + 1e-5f);
    for (int i = threadIdx.x; i < P; i += blockDim.x) {
        float v = (xi[i] - mean) * inv;
        if (doLeaky) v = (v >= 0.f) ? v : 0.2f * v;
        yo[i] = v;
    }
}

// ---------------- strided channel-offset copy with scale / leaky ----------------
__global__ void copych_kernel(const float* __restrict__ x, float* __restrict__ y,
                              int C, int P, int coOff, int Ctot,
                              float scale, int doLeaky, size_t total)
{
    size_t idx = (size_t)blockIdx.x * blockDim.x + threadIdx.x;
    if (idx >= total) return;
    int p = idx % P; size_t t = idx / P;
    int c = t % C;
    int b = t / C;
    float v = x[idx] * scale;
    if (doLeaky) v = (v >= 0.f) ? v : 0.2f * v;
    y[((size_t)b * Ctot + coOff + c) * P + p] = v;
}

// ---------------- zero pad [BC,2,2,2] -> [BC,3,3,3] (pl=0, pr=1 per dim) ----------------
__global__ void pad222_kernel(const float* __restrict__ x, float* __restrict__ y, int BC)
{
    int idx = blockIdx.x * blockDim.x + threadIdx.x;
    int total = BC * 27;
    if (idx >= total) return;
    int p = idx % 27; int bc = idx / 27;
    int d = p % 3; int t = p / 3; int w = t % 3; int h = t / 3;
    float v = 0.f;
    if (h < 2 && w < 2 && d < 2) v = x[(size_t)bc * 8 + (h * 2 + w) * 2 + d];
    y[idx] = v;
}

// ---------------- SASA (faithful to the memory-reinterpreting reshape) ----------------
__global__ void sasa_kernel(const float* __restrict__ q, const float* __restrict__ kp,
                            const float* __restrict__ vp, const float* __restrict__ rh,
                            const float* __restrict__ rw, const float* __restrict__ rd,
                            float* __restrict__ out, int C, int Chalf)
{
    const int NT = 256;
    int bb  = blockIdx.x >> 3;   // batch
    int blk = blockIdx.x & 7;    // channel block == softmax "position"
    const int CB = C / 8;
    __shared__ float red[15][NT];
    __shared__ float wgt[8];

    float dot[8] = {0, 0, 0, 0, 0, 0, 0, 0};
    float qs = 0.f;
    for (int i = threadIdx.x; i < CB * 8; i += NT) {
        int c = blk * CB + (i >> 3);
        int s = i & 7;
        int h = (s >> 2) & 1, w = (s >> 1) & 1, d = s & 1;
        float qv = q[((size_t)bb * C + c) * 8 + s];
        qs += qv;
        size_t kbase = ((size_t)bb * C + c) * 27;
#pragma unroll
        for (int n = 0; n < 8; ++n) {
            int ni = (n >> 2) & 1, nj = (n >> 1) & 1, nl = n & 1;
            dot[n] += qv * kp[kbase + ((h + ni) * 3 + (w + nj)) * 3 + (d + nl)];
        }
    }
    float relp[6] = {0, 0, 0, 0, 0, 0};
    for (int c2 = threadIdx.x; c2 < Chalf; c2 += NT) {
        relp[0] += rh[2 * c2]; relp[1] += rh[2 * c2 + 1];
        relp[2] += rw[2 * c2]; relp[3] += rw[2 * c2 + 1];
        relp[4] += rd[2 * c2]; relp[5] += rd[2 * c2 + 1];
    }
#pragma unroll
    for (int n = 0; n < 8; ++n) red[n][threadIdx.x] = dot[n];
    red[8][threadIdx.x] = qs;
#pragma unroll
    for (int n = 0; n < 6; ++n) red[9 + n][threadIdx.x] = relp[n];
    __syncthreads();
    for (int o = NT / 2; o > 0; o >>= 1) {
        if ((int)threadIdx.x < o) {
#pragma unroll
            for (int n = 0; n < 15; ++n) red[n][threadIdx.x] += red[n][threadIdx.x + o];
        }
        __syncthreads();
    }
    if (threadIdx.x == 0) {
        float qsum = red[8][0];
        float e[8]; float mx = -1e30f;
#pragma unroll
        for (int n = 0; n < 8; ++n) {
            int ni = (n >> 2) & 1, nj = (n >> 1) & 1, nl = n & 1;
            float rel = red[9 + ni][0] + red[11 + nj][0] + red[13 + nl][0];
            e[n] = red[n][0] + qsum * rel;
            mx = fmaxf(mx, e[n]);
        }
        float sum = 0.f;
#pragma unroll
        for (int n = 0; n < 8; ++n) { e[n] = __expf(e[n] - mx); sum += e[n]; }
#pragma unroll
        for (int n = 0; n < 8; ++n) wgt[n] = e[n] / sum;
    }
    __syncthreads();
    for (int i = threadIdx.x; i < CB * 8; i += NT) {
        int c = blk * CB + (i >> 3);
        int s = i & 7;
        int h = (s >> 2) & 1, w = (s >> 1) & 1, d = s & 1;
        size_t vbase = ((size_t)bb * C + c) * 27;
        float a2 = 0.f;
#pragma unroll
        for (int n = 0; n < 8; ++n) {
            int ni = (n >> 2) & 1, nj = (n >> 1) & 1, nl = n & 1;
            a2 += wgt[n] * vp[vbase + ((h + ni) * 3 + (w + nj)) * 3 + (d + nl)];
        }
        size_t oi = ((size_t)bb * C + c) * 8 + s;
        out[oi] = out[oi] + a2;   // accumulate onto resconv+bn result
    }
}

// =====================================================================
// Host orchestration
// =====================================================================
extern "C" void kernel_launch(void* const* d_in, const int* in_sizes, int n_in,
                              void* d_out, int out_size, void* d_ws, size_t ws_size,
                              hipStream_t stream)
{
    (void)in_sizes; (void)n_in; (void)out_size;
    const int B = 8;
    const float BN_SCALE = 1.0f / sqrtf(1.0f + 1e-5f);

    // ---- parameter unpack (setup_inputs dict order) ----
    int ix = 0;
    auto F = [&](int i) { return (const float*)d_in[i]; };
    const float* X1 = F(ix++);
    const float* X2 = F(ix++);
    struct RC { const float *dww, *dwb, *pww, *pwb; };
    auto getRC = [&]() { RC r; r.dww = F(ix++); r.dwb = F(ix++); r.pww = F(ix++); r.pwb = F(ix++); return r; };
    RC conv1[3], conv2[3];
    for (int i = 0; i < 3; ++i) conv1[i] = getRC();
    for (int i = 0; i < 3; ++i) conv2[i] = getRC();
    struct BT { const float *c1, *c2, *c3; };
    auto getBT = [&]() { BT r; r.c1 = F(ix++); r.c2 = F(ix++); r.c3 = F(ix++); return r; };
    BT rb[4][2];
    for (int s = 0; s < 4; ++s) { rb[s][0] = getBT(); rb[s][1] = getBT(); }
    RC sc[4][2]; int scn[4] = {2, 2, 2, 1};
    for (int s = 0; s < 4; ++s) for (int i = 0; i < scn[s]; ++i) sc[s][i] = getRC();
    struct SA { RC r1, r2; const float *kw, *kb, *qw, *qb, *vw, *vb, *rh, *rw, *rd; };
    auto getSA = [&]() {
        SA a; a.r1 = getRC(); a.r2 = getRC();
        a.kw = F(ix++); a.kb = F(ix++); a.qw = F(ix++); a.qb = F(ix++);
        a.vw = F(ix++); a.vb = F(ix++);
        a.rh = F(ix++); a.rw = F(ix++); a.rd = F(ix++); return a;
    };
    SA att1 = getSA(), att2 = getSA();
    const float* cls1w = F(ix++); const float* cls1b = F(ix++);
    const float* cls2w = F(ix++); const float* cls2b = F(ix++);
    const float* fc1w  = F(ix++); const float* fc1b  = F(ix++);
    const float* fc2w  = F(ix++); const float* fc2b  = F(ix++);

    // ---- workspace carve (floats): 17M big + 5 x 9M = 62M floats (~248 MB) ----
    const size_t M1 = 1u << 20;
    size_t needF = 62 * M1;
    if (ws_size < needF * sizeof(float)) return;   // workspace too small; no-op
    float* ws = (float*)d_ws;
    size_t off = 0;
    auto carve = [&](size_t n) { float* p = ws + off; off += n; return p; };
    float* big0 = carve(17 * M1);
    float* s2 = carve(9 * M1);
    float* s3 = carve(9 * M1);
    float* s4 = carve(9 * M1);
    float* s5 = carve(9 * M1);
    float* s6 = carve(9 * M1);
    float* bigB = big0 + (17 * M1) / 2;   // second half of big slot (>= 8.4M floats each)

    // ---- launch helpers ----
    auto cdiv = [](long long a, long long b) { return (unsigned)((a + b - 1) / b); };
    auto gemm = [&](const float* Wt, const float* bias, const float* X, float* Y,
                    int Cout, int Cin, int ks, int pad, int Bn, int D, int H, int Wd,
                    int coOff, int Ctot, float scale, int leak) {
        int P = D * H * Wd;
        int tilesN = (Bn * P + 15) / 16;
        if (Cout >= 256) {   // 8-wave block, LDS-shared B operand
            int tilesMB = (Cout + 255) / 256;
            dim3 grid((unsigned)(tilesMB * tilesN));
            if (ks == 1)
                gemm_wmma_lds_kernel<1><<<grid, 256, 0, stream>>>(Wt, bias, X, Y, Cout, Cin, pad,
                                                                  Bn, D, H, Wd, tilesMB, coOff, Ctot, scale, leak);
            else
                gemm_wmma_lds_kernel<3><<<grid, 256, 0, stream>>>(Wt, bias, X, Y, Cout, Cin, pad,
                                                                  Bn, D, H, Wd, tilesMB, coOff, Ctot, scale, leak);
            return;
        }
        const bool mt2 = Cout >= 32;
        const int mtv = mt2 ? 2 : 1;
        int tilesM = (Cout + 16 * mtv - 1) / (16 * mtv);
        dim3 grid((unsigned)(tilesM * tilesN));
        if (ks == 1) {
            if (mt2) gemm_wmma_kernel<1, 2><<<grid, 32, 0, stream>>>(Wt, bias, X, Y, Cout, Cin, pad,
                                                                     Bn, D, H, Wd, tilesM, coOff, Ctot, scale, leak);
            else     gemm_wmma_kernel<1, 1><<<grid, 32, 0, stream>>>(Wt, bias, X, Y, Cout, Cin, pad,
                                                                     Bn, D, H, Wd, tilesM, coOff, Ctot, scale, leak);
        } else {
            if (mt2) gemm_wmma_kernel<3, 2><<<grid, 32, 0, stream>>>(Wt, bias, X, Y, Cout, Cin, pad,
                                                                     Bn, D, H, Wd, tilesM, coOff, Ctot, scale, leak);
            else     gemm_wmma_kernel<3, 1><<<grid, 32, 0, stream>>>(Wt, bias, X, Y, Cout, Cin, pad,
                                                                     Bn, D, H, Wd, tilesM, coOff, Ctot, scale, leak);
        }
    };
    auto dwc = [&](const float* X, const RC& r, float* Y, int Bn, int C, int D, int H, int Wd) {
        long long total = (long long)Bn * C * D * H * Wd;
        dwconv3_kernel<<<cdiv(total, 256), 256, 0, stream>>>(X, r.dww, r.dwb, Y, Bn, C, D, H, Wd);
    };
    auto maxpool = [&](const float* X, float* Y, int BC, int D, int H, int Wd) {
        long long total = (long long)BC * (D / 2) * (H / 2) * (Wd / 2);
        maxpool2_kernel<<<cdiv(total, 256), 256, 0, stream>>>(X, Y, BC, D, H, Wd);
    };
    auto inorm = [&](const float* X, float* Y, int C, int P, int coOff, int Ctot, int leak) {
        inorm_kernel<<<B * C, 256, 0, stream>>>(X, Y, C, P, coOff, Ctot, leak);
    };
    auto copych = [&](const float* X, float* Y, int C, int P, int coOff, int Ctot,
                      float scale, int leak) {
        long long total = (long long)B * C * P;
        copych_kernel<<<cdiv(total, 256), 256, 0, stream>>>(X, Y, C, P, coOff, Ctot,
                                                            scale, leak, (size_t)total);
    };

    // ---- phase 1: conv_1 / conv_2 chains at 64^3 then maxpool -> 32^3, 8ch ----
    auto phase1 = [&](const float* Xin, RC* cv, float* outp) {
        int chans[4] = {1, 2, 4, 8};
        float* pong[3] = {s5, s6, big0};
        const float* cur = Xin;
        for (int i = 0; i < 3; ++i) {
            dwc(cur, cv[i], s2, B, chans[i], 64, 64, 64);
            gemm(cv[i].pww, cv[i].pwb, s2, pong[i], chans[i + 1], chans[i], 1, 0,
                 B, 64, 64, 64, 0, chans[i + 1], 1.f, 0);
            cur = pong[i];
        }
        maxpool(cur, outp, B * 8, 64, 64, 64);
    };
    phase1(X1, conv1, s3);
    phase1(X2, conv2, s4);

    // ---- bottleneck helper ----
    auto bottleneck = [&](const BT& p, const float* xa, const float* xb, float* Aout,
                          int c, int n) {
        int P = n * n * n;
        gemm(p.c1, nullptr, xa, s5, c, c, 1, 0, B, n, n, n, 0, c, 1.f, 0);
        inorm(s5, s6, c, P, 0, c, 1);
        gemm(p.c2, nullptr, s6, s5, c, c, 3, 1, B, n, n, n, 0, c, 1.f, 0);
        inorm(s5, s6, c, P, 0, c, 1);
        gemm(p.c3, nullptr, s6, s5, 3 * c, c, 1, 0, B, n, n, n, 0, 3 * c, 1.f, 0);
        inorm(s5, Aout, 3 * c, P, 0, 4 * c, 1);          // leaky fused (post-concat)
        copych(xb, Aout, c, P, 3 * c, 4 * c, 1.f, 1);    // concat tail with leaky
    };

    // ---- stages 1..4 ----
    int cs[4] = {8, 32, 128, 512};
    int ds[4] = {32, 16, 8, 4};
    for (int st = 0; st < 4; ++st) {
        int c = cs[st], n = ds[st], C4 = 4 * c;
        bottleneck(rb[st][0], s3, s4, big0, c, n);
        bottleneck(rb[st][1], s4, s3, bigB, c, n);
        float* a = big0; float* ap = s5;
        float* bq = bigB; float* bp = s6;
        for (int i = 0; i < scn[st]; ++i) {
            int last = (i == scn[st] - 1);
            float sc_s = last ? BN_SCALE : 1.f;
            int   sc_l = last ? 1 : 0;
            dwc(a, sc[st][i], s2, B, C4, n, n, n);
            gemm(sc[st][i].pww, sc[st][i].pwb, s2, ap, C4, C4, 1, 0, B, n, n, n, 0, C4, sc_s, sc_l);
            { float* t = a; a = ap; ap = t; }
            dwc(bq, sc[st][i], s2, B, C4, n, n, n);
            gemm(sc[st][i].pww, sc[st][i].pwb, s2, bp, C4, C4, 1, 0, B, n, n, n, 0, C4, sc_s, sc_l);
            { float* t = bq; bq = bp; bp = t; }
        }
        maxpool(a,  s3, B * C4, n, n, n);
        maxpool(bq, s4, B * C4, n, n, n);
    }
    // now: x1 = s3, x2 = s4, each [8, 2048, 2,2,2]

    // ---- shared attention (y = bn(resconv(resconv(xa))) + sasa(xb)) ----
    auto sharedatt = [&](const SA& A, const float* xa, const float* xb, float* yout) {
        dwc(xa, A.r1, s2, B, 2048, 2, 2, 2);
        gemm(A.r1.pww, A.r1.pwb, s2, s5, 2048, 2048, 1, 0, B, 2, 2, 2, 0, 2048, 1.f, 0);
        dwc(s5, A.r2, s2, B, 2048, 2, 2, 2);
        gemm(A.r2.pww, A.r2.pwb, s2, yout, 2048, 2048, 1, 0, B, 2, 2, 2, 0, 2048, BN_SCALE, 0);
        // sasa on xb
        float* padb = s2;
        float* kbuf = s2 + 1 * M1;
        float* vbuf = s2 + 2 * M1;
        float* qbuf = s2 + 3 * M1;
        pad222_kernel<<<cdiv((long long)B * 2048 * 27, 256), 256, 0, stream>>>(xb, padb, B * 2048);
        gemm(A.kw, A.kb, padb, kbuf, 2048, 2048, 1, 0, B, 3, 3, 3, 0, 2048, 1.f, 0);
        gemm(A.vw, A.vb, padb, vbuf, 2048, 2048, 1, 0, B, 3, 3, 3, 0, 2048, 1.f, 0);
        gemm(A.qw, A.qb, xb,   qbuf, 2048, 2048, 1, 0, B, 2, 2, 2, 0, 2048, 1.f, 0);
        sasa_kernel<<<B * 8, 256, 0, stream>>>(qbuf, kbuf, vbuf, A.rh, A.rw, A.rd, yout, 2048, 1024);
    };
    float* y1 = big0;
    float* y2 = bigB;
    sharedatt(att1, s3, s4, y1);
    sharedatt(att2, s4, s3, y2);

    // ---- heads: Linear+BN+Leaky (as KS=1 GEMM, P=1, column=batch), concat into h ----
    float* h = s5;   // [8, 2600]
    gemm(cls1w, cls1b, y1, h, 1300, 16384, 1, 0, B, 1, 1, 1, 0,    2600, BN_SCALE, 1);
    gemm(cls2w, cls2b, y2, h, 1300, 16384, 1, 0, B, 1, 1, 1, 1300, 2600, BN_SCALE, 1);
    gemm(fc1w, fc1b, h, s6, 500, 2600, 1, 0, B, 1, 1, 1, 0, 500, BN_SCALE, 1);
    gemm(fc2w, fc2b, s6, (float*)d_out, 2, 500, 1, 0, B, 1, 1, 1, 0, 2, 1.f, 0);
}
